// FasterRCNN_12154757447763
// MI455X (gfx1250) — matrix operand order repaired and checked by reading, not compiled, not run on hardware
//
#include <hip/hip_runtime.h>
#include <math.h>

// ---------------------------------------------------------------------------
// Types / helpers
// ---------------------------------------------------------------------------
typedef unsigned int  u32;
typedef unsigned short u16;
typedef __attribute__((ext_vector_type(16))) __bf16 v16bf;
typedef __attribute__((ext_vector_type(8)))  float  v8f;

union ABuf { u32 u[8]; v16bf v; };

// gfx1250 async global->LDS copy path (ASYNCcnt). Guarded so the build is safe
// on toolchains that do not declare the builtin. Probe round 2 showed the
// prototype is (int4 AS1* src, int4 AS3* dst, imm offset, imm cpol).
#if defined(__AMDGCN__) && __has_builtin(__builtin_amdgcn_global_load_async_to_lds_b128)
#define HAVE_ASYNC_LDS 1
typedef int v4i_t __attribute__((vector_size(16)));
typedef __attribute__((address_space(1))) v4i_t as1_v4i;
typedef __attribute__((address_space(3))) v4i_t as3_v4i;
#else
#define HAVE_ASYNC_LDS 0
#endif

__device__ __forceinline__ void wait_async_lds() {
#if HAVE_ASYNC_LDS
#if __has_builtin(__builtin_amdgcn_s_wait_asynccnt)
  __builtin_amdgcn_s_wait_asynccnt(0);
#else
  asm volatile("s_wait_asynccnt 0" ::: "memory");
#endif
#endif
}

__device__ __forceinline__ u16 f2bf(float f) {
  u32 u = __float_as_uint(f);
  u32 r = u + 0x7FFFu + ((u >> 16) & 1u);     // RNE
  return (u16)(r >> 16);
}
__device__ __forceinline__ u32 pack2(float a, float b) {
  return (u32)f2bf(a) | ((u32)f2bf(b) << 16);
}
__device__ __forceinline__ u32 fmono(float f) {  // monotone float->uint
  u32 u = __float_as_uint(f);
  return u ^ ((u >> 31) ? 0xFFFFFFFFu : 0x80000000u);
}
__device__ __forceinline__ float fsig(float x) { return 1.f / (1.f + expf(-x)); }

__device__ __forceinline__ float iou_box(float4 a, float4 b) {
  float aA = (a.z - a.x) * (a.w - a.y);
  float aB = (b.z - b.x) * (b.w - b.y);
  float lx = fmaxf(a.x, b.x), ly = fmaxf(a.y, b.y);
  float rx = fminf(a.z, b.z), ry = fminf(a.w, b.w);
  float w = fmaxf(rx - lx, 0.f), h = fmaxf(ry - ly, 0.f);
  float inter = w * h;
  return inter / fmaxf(aA + aB - inter, 1e-9f);
}

#define BBOX_CLAMP 4.135166556742356f
#define SEL_CAP 16384

// pyramid geometry
__constant__ int c_H[5]     = {128, 64, 32, 16, 8};
__constant__ int c_W[5]     = {208, 104, 52, 26, 13};
__constant__ int c_HW[5]    = {26624, 6656, 1664, 416, 104};
__constant__ int c_cumHW[6] = {0, 26624, 33280, 34944, 35360, 35464};
__constant__ int c_STRIDE[5]= {4, 8, 16, 32, 64};
__constant__ int c_TILES[5] = {416, 104, 26, 7, 2};     // ceil(HW/64)
__constant__ int c_cumK[6]  = {0, 1000, 2000, 3000, 4000, 4312};

#define HWSUM   35464
#define LOGN    106392      // 3*HWSUM per image
#define DTOT    4312        // selected per image
#define NSEL    8624        // 2*DTOT
#define NPROP   2000
#define FEATK   12544       // 256*49

// ---------------------------------------------------------------------------
// Packing kernels (fp32 -> bf16 pair layouts)
// ---------------------------------------------------------------------------
__global__ void k_pack_fmap(const float* f0, const float* f1, const float* f2,
                            const float* f3, const float* f4, u32* out, int total) {
  int i = blockIdx.x * blockDim.x + threadIdx.x;
  if (i >= total) return;
  int lvl = 0;
  while (lvl < 4 && i >= 256 * c_cumHW[lvl + 1]) ++lvl;
  int r  = i - 256 * c_cumHW[lvl];
  int HW = c_HW[lvl];
  int img = r / (128 * HW);
  int rr  = r - img * 128 * HW;
  int pr  = rr / HW;
  int p   = rr - pr * HW;
  const float* fm = lvl == 0 ? f0 : lvl == 1 ? f1 : lvl == 2 ? f2 : lvl == 3 ? f3 : f4;
  size_t base = ((size_t)img * 256 + 2 * pr) * HW + p;
  out[i] = pack2(fm[base], fm[base + HW]);
}

// conv weights -> [tap][ciChunk(8)][co(256)][16 pairs]
__global__ void k_pack_convw(const float* w, u32* out) {
  int i = blockIdx.x * blockDim.x + threadIdx.x;
  if (i >= 9 * 8 * 256 * 16) return;
  int j  = i & 15;
  int co = (i >> 4) & 255;
  int cc = (i >> 12) & 7;
  int tap = i >> 15;
  int ci = cc * 32 + 2 * j;
  size_t s0 = ((size_t)co * 256 + ci) * 9 + tap;
  out[i] = pack2(w[s0], w[s0 + 9]);
}

// generic row-major [N][K] fp32 -> [Npad][K] bf16 (zero rows beyond N)
__global__ void k_pack_mat(const float* src, u16* dst, int N, int K, int Npad) {
  int i = blockIdx.x * blockDim.x + threadIdx.x;
  if (i >= Npad * K) return;
  int n = i / K, k = i - n * K;
  dst[i] = (n < N) ? f2bf(src[(size_t)n * K + k]) : (u16)0;
}

__global__ void k_zero(u32* p, int n) {
  int i = blockIdx.x * blockDim.x + threadIdx.x;
  if (i < n) p[i] = 0u;
}

// ---------------------------------------------------------------------------
// RPN: fused 3x3 conv (256->256, relu) + 1x1 log/reg heads, WMMA bf16.
// Block: 256 thr (8 waves). Tile: 256 co x 64 positions. K-loop 9 taps * 8
// ci-chunks of 32. LDS is double-buffered: the next K-step's 16KB weight tile
// is fetched with GLOBAL_LOAD_ASYNC_TO_LDS_B128 (ASYNCcnt) while the current
// step's WMMAs execute -> one barrier per K-step. Heads are computed from the
// relu activation held in LDS so the 256ch activation never touches HBM.
// ---------------------------------------------------------------------------
__global__ void __launch_bounds__(256) k_rpn_conv(
    const u32* __restrict__ fmapP, const u32* __restrict__ convWp,
    const float* __restrict__ conv_b, const float* __restrict__ log_w,
    const float* __restrict__ log_b, const float* __restrict__ reg_w,
    const float* __restrict__ reg_b, float* __restrict__ logits,
    float* __restrict__ regs) {
  extern __shared__ char smem[];
  u32*   As = (u32*)smem;              // 2 x (256 co * 16 pairs)
  u32*   Bs = As + 2 * 4096;           // 2 x (64 pos * 16 pairs)
  float* Xs = (float*)(Bs + 2 * 1024); // 64 pos * 256 co (relu act)
  float* Wh = Xs + 64 * 256;           // 15 * 256 head weights
  float* Bc = Wh + 15 * 256;           // 256 conv bias

  int tid = threadIdx.x;
  int bid = blockIdx.x;
  int img = bid / 555;
  int r = bid % 555;
  int lvl = 0;
  while (r >= c_TILES[lvl]) { r -= c_TILES[lvl]; ++lvl; }
  int H = c_H[lvl], W = c_W[lvl], HW = c_HW[lvl];
  int posBase = r * 64;
  const u32* fp = fmapP + (size_t)256 * c_cumHW[lvl];

  for (int i = tid; i < 15 * 256; i += 256) {
    int h = i >> 8, c = i & 255;
    Wh[i] = (h < 3) ? log_w[h * 256 + c] : reg_w[(h - 3) * 256 + c];
  }
  if (tid < 256) Bc[tid] = conv_b[tid];

  int lane = tid & 31, wv = tid >> 5;
  int half = lane >> 4, l15 = lane & 15;

  v8f acc[2][4];
  for (int a = 0; a < 2; ++a)
    for (int b = 0; b < 4; ++b)
      for (int e = 0; e < 8; ++e) acc[a][b][e] = 0.f;

  for (int kk = 0; kk <= 72; ++kk) {
    // ---- stage K-step kk into buffer kk&1 (overlapped with compute below) --
    if (kk < 72) {
      int tap = kk >> 3, cc = kk & 7;
      int ky = tap / 3 - 1, kx = tap % 3 - 1;
      int buf = kk & 1;
      u32* Ad = As + buf * 4096;
      const u32* aw = convWp + (size_t)(tap * 8 + cc) * 4096;
#if HAVE_ASYNC_LDS
      {
        as1_v4i* g = (as1_v4i*)(u32*)aw;   // 16B chunks, global AS
        as3_v4i* l = (as3_v4i*)(void*)Ad;  // 16B chunks, LDS AS
#pragma unroll
        for (int rnd = 0; rnd < 4; ++rnd) {
          int idx = tid + rnd * 256;       // 1024 x 16B = 16KB tile
          __builtin_amdgcn_global_load_async_to_lds_b128(g + idx, l + idx, 0, 0);
        }
      }
#else
      for (int i = tid; i < 4096; i += 256) Ad[i] = aw[i];
#endif
      u32* Bd = Bs + buf * 1024;
      for (int i = tid; i < 1024; i += 256) {
        int pos = i >> 4, j = i & 15;
        int p = posBase + pos;
        u32 v = 0u;
        if (p < HW) {
          int y = p / W + ky, x = p % W + kx;
          if (y >= 0 && y < H && x >= 0 && x < W)
            v = fp[((size_t)(img * 128 + cc * 16 + j) * H + y) * W + x];
        }
        Bd[pos * 16 + j] = v;
      }
    }
    // ---- compute K-step kk-1 from buffer (kk-1)&1 ----
    if (kk > 0) {
      int buf = (kk - 1) & 1;
      const u32* Ar = As + buf * 4096;
      const u32* Br = Bs + buf * 1024;
      ABuf ta0, ta1;
      const u32* ap0 = Ar + (wv * 32 + l15) * 16;
      const u32* ap1 = Ar + (wv * 32 + 16 + l15) * 16;
#pragma unroll
      for (int v = 0; v < 8; ++v) {
        int pi = ((v >> 2) << 3) + half * 4 + (v & 3);   // ISA 16-bit A layout
        ta0.u[v] = ap0[pi];
        ta1.u[v] = ap1[pi];
      }
#pragma unroll
      for (int s = 0; s < 4; ++s) {
        ABuf tb;
        const u32* bp = Br + (s * 16 + l15) * 16;
#pragma unroll
        for (int v = 0; v < 8; ++v) tb.u[v] = bp[((v >> 2) << 3) + half * 4 + (v & 3)];  // ISA B layout
        acc[0][s] = __builtin_amdgcn_wmma_f32_16x16x32_bf16(false, ta0.v, false, tb.v,
                                                            (short)0, acc[0][s], false, false);
        acc[1][s] = __builtin_amdgcn_wmma_f32_16x16x32_bf16(false, ta1.v, false, tb.v,
                                                            (short)0, acc[1][s], false, false);
      }
    }
    wait_async_lds();   // own async-to-LDS issues complete before the barrier
    __syncthreads();
  }

  for (int cs = 0; cs < 2; ++cs)
    for (int s = 0; s < 4; ++s)
#pragma unroll
      for (int v = 0; v < 8; ++v) {
        int co  = wv * 32 + cs * 16 + v + 8 * half;   // D-tile: m = vgpr + 8*(lane>=16)
        int pos = s * 16 + l15;                       //         n = lane&15
        float x = acc[cs][s][v] + Bc[co];
        Xs[pos * 256 + co] = x > 0.f ? x : 0.f;
      }
  __syncthreads();
  int off3 = 3 * c_cumHW[lvl];
  for (int o = tid; o < 64 * 15; o += 256) {
    int pos = o / 15, h = o % 15;
    int p = posBase + pos;
    if (p >= HW) continue;
    const float* w  = Wh + h * 256;
    const float* xr = Xs + pos * 256;
    float s = (h < 3) ? log_b[h] : reg_b[h - 3];
    for (int c = 0; c < 256; ++c) s += w[c] * xr[c];
    if (h < 3) {
      logits[(size_t)img * LOGN + off3 + p * 3 + h] = s;
    } else {
      int a = (h - 3) >> 2, cmp = (h - 3) & 3;
      regs[((size_t)img * LOGN + off3 + p * 3 + a) * 4 + cmp] = s;
    }
  }
}

// ---------------------------------------------------------------------------
// Generic bf16 GEMM:  C[M][Npad] = A[M][K] * W[Npad][K]^T (+bias, relu).
// One wave = 16M x 64N, fragments straight from L2 (all operands L2-resident),
// with global_prefetch of the next K-chunk.
// ---------------------------------------------------------------------------
__global__ void __launch_bounds__(128) k_gemm(
    const u16* __restrict__ A, const u16* __restrict__ Wt,
    const float* __restrict__ bias, int M, int K, int Npad, int Nbias,
    int relu, float* __restrict__ outF, u16* __restrict__ outB) {
  int gw = (blockIdx.x * blockDim.x + threadIdx.x) >> 5;
  int lane = threadIdx.x & 31;
  int Mt = M >> 4;
  int Ng = (Npad + 63) >> 6;
  if (gw >= Mt * Ng) return;
  int mt = gw % Mt, ng = gw / Mt;
  int m0 = mt << 4, n0 = ng << 6;
  int half = lane >> 4, l15 = lane & 15;
  int Kp = K >> 1;
  const u32* Arow = (const u32*)A + (size_t)(m0 + l15) * Kp;
  const u32* Brow0 = (const u32*)Wt + (size_t)(n0 + l15) * Kp;
  v8f acc[4];
  for (int i = 0; i < 4; ++i)
    for (int e = 0; e < 8; ++e) acc[i][e] = 0.f;
  for (int k0 = 0; k0 < K; k0 += 32) {
    int kb = k0 >> 1;
    if (k0 + 32 < K) {   // hide L2 latency for next chunk (global_prefetch_b8)
      __builtin_prefetch((const void*)(Arow + kb + 16), 0, 0);
      __builtin_prefetch((const void*)(Brow0 + kb + 16), 0, 0);
    }
    ABuf ta;
#pragma unroll
    for (int v = 0; v < 8; ++v)
      ta.u[v] = Arow[kb + ((v >> 2) << 3) + half * 4 + (v & 3)];
#pragma unroll
    for (int nt = 0; nt < 4; ++nt) {
      int nb = n0 + (nt << 4);
      if (nb < Npad) {
        const u32* Brow = (const u32*)Wt + (size_t)(nb + l15) * Kp;
        ABuf tb;
#pragma unroll
        for (int v = 0; v < 8; ++v) tb.u[v] = Brow[kb + ((v >> 2) << 3) + half * 4 + (v & 3)];
        acc[nt] = __builtin_amdgcn_wmma_f32_16x16x32_bf16(false, ta.v, false, tb.v,
                                                          (short)0, acc[nt], false, false);
      }
    }
  }
  for (int nt = 0; nt < 4; ++nt) {
    int nb = n0 + (nt << 4);
    if (nb >= Npad) continue;
    int n = nb + l15;
    float b = (bias && n < Nbias) ? bias[n] : 0.f;
#pragma unroll
    for (int v = 0; v < 8; ++v) {
      int m = m0 + v + 8 * half;
      float x = acc[nt][v] + b;
      if (relu && x < 0.f) x = 0.f;
      if (outF) outF[(size_t)m * Npad + n] = x;
      if (outB) outB[(size_t)m * Npad + n] = f2bf(x);
    }
  }
}

// ---------------------------------------------------------------------------
// Histogram top-k selection: hist -> threshold bin -> compact -> bitonic sort
// ---------------------------------------------------------------------------
__global__ void k_hist(const float* vals, int n, u32* hist) {
  int i = blockIdx.x * blockDim.x + threadIdx.x;
  if (i >= n) return;
  atomicAdd(&hist[fmono(vals[i]) >> 22], 1u);
}

__global__ void k_thresh(const u32* hist, int k, u32* cnt) {
  if (threadIdx.x || blockIdx.x) return;
  u32 acc = 0;
  int b = 1023;
  for (; b >= 0; --b) { acc += hist[b]; if ((int)acc >= k) break; }
  if (b < 0) b = 0;
  cnt[1] = (u32)b;
}

__global__ void k_compact(const float* __restrict__ vals, int n, u32* cnt,
                          float* __restrict__ cv, u32* __restrict__ ci, int mode) {
  int i = blockIdx.x * blockDim.x + threadIdx.x;
  if (i >= n) return;
  u32 bs = cnt[1];
  float v = vals[i];
  u32 b = fmono(v) >> 22;
  bool sel = mode ? (b > bs) : (b == bs);
  if (sel) {
    u32 pos = atomicAdd(&cnt[0], 1u);
    if (pos < (u32)SEL_CAP) { cv[pos] = v; ci[pos] = (u32)i; }
  }
}

// single-workgroup 16K bitonic sort (desc, tie-break idx asc) in 128KB LDS
__global__ void __launch_bounds__(1024) k_sort16k(
    const float* __restrict__ val_in, const u32* __restrict__ idx_in,
    const u32* cntPtr, int cntFixed, float* __restrict__ val_out,
    u32* __restrict__ idx_out) {
  extern __shared__ char smem[];
  float* sv = (float*)smem;
  u32*   si = (u32*)(sv + SEL_CAP);
  int n = cntFixed;
  if (cntPtr) { u32 c = *cntPtr; n = (c > (u32)SEL_CAP) ? SEL_CAP : (int)c; }
  for (int i = threadIdx.x; i < SEL_CAP; i += blockDim.x) {
    if (i < n) { sv[i] = val_in[i]; si[i] = idx_in ? idx_in[i] : (u32)i; }
    else       { sv[i] = -3.4e38f;  si[i] = 0x7FFFFFFFu; }
  }
  __syncthreads();
  for (int k = 2; k <= SEL_CAP; k <<= 1)
    for (int j = k >> 1; j > 0; j >>= 1) {
      for (int i = threadIdx.x; i < SEL_CAP; i += blockDim.x) {
        int p = i ^ j;
        if (p > i) {
          float a = sv[i], b = sv[p];
          u32 ia = si[i], ib = si[p];
          bool desc = ((i & k) == 0);
          bool good = (a > b) || (a == b && ia < ib);
          if (desc ? !good : good) { sv[i] = b; sv[p] = a; si[i] = ib; si[p] = ia; }
        }
      }
      __syncthreads();
    }
  for (int i = threadIdx.x; i < SEL_CAP; i += blockDim.x) {
    val_out[i] = sv[i];
    idx_out[i] = si[i];
  }
}

__global__ void k_take(const float* sv, const u32* si, int k, int nLimit,
                       u32* outIdx, float* outVal) {
  int i = blockIdx.x * blockDim.x + threadIdx.x;
  if (i >= k) return;
  u32 id = si[i];
  float v = sv[i];
  if (id >= (u32)nLimit) { id = 0; v = -3.4e38f; }
  outIdx[i] = id;
  if (outVal) outVal[i] = v;
}

// ---------------------------------------------------------------------------
// Stage-1 decode / sort / NMS / proposal compaction
// ---------------------------------------------------------------------------
__global__ void k_decode1(const float* __restrict__ logits, const float* __restrict__ regs,
                          const u32* __restrict__ sel, const float* p0, const float* p1,
                          const float* p2, const float* p3, const float* p4,
                          const int* imsizes, float4* dbox, float* dscore,
                          float* dgoff, u32* dimg) {
  int s = blockIdx.x * blockDim.x + threadIdx.x;
  if (s >= NSEL) return;
  int img = s / DTOT, rem = s - img * DTOT;
  int lvl = 0;
  while (lvl < 4 && rem >= c_cumK[lvl + 1]) ++lvl;
  const float* pr = lvl == 0 ? p0 : lvl == 1 ? p1 : lvl == 2 ? p2 : lvl == 3 ? p3 : p4;
  int ti = (int)sel[s];
  int off3 = 3 * c_cumHW[lvl];
  float l = logits[(size_t)img * LOGN + off3 + ti];
  float score = fsig(l);
  const float* pp = pr + (size_t)ti * 4;
  const float* rg = regs + ((size_t)img * LOGN + off3 + ti) * 4;
  float w = pp[2] - pp[0], h = pp[3] - pp[1];
  float cx = pp[0] + 0.5f * w + rg[0] * w;
  float cy = pp[1] + 0.5f * h + rg[1] * h;
  float ww = w * expf(fminf(rg[2], BBOX_CLAMP));
  float hh = h * expf(fminf(rg[3], BBOX_CLAMP));
  float Wc = (float)imsizes[img * 2 + 1], Hc = (float)imsizes[img * 2 + 0];
  float x1 = fminf(fmaxf(cx - 0.5f * ww, 0.f), Wc);
  float x2 = fminf(fmaxf(cx + 0.5f * ww, 0.f), Wc);
  float y1 = fminf(fmaxf(cy - 0.5f * hh, 0.f), Hc);
  float y2 = fminf(fmaxf(cy + 0.5f * hh, 0.f), Hc);
  dbox[s] = make_float4(x1, y1, x2, y2);
  dscore[s] = score;
  dgoff[s] = (float)(img * 10 + lvl) * 4096.f;
  dimg[s] = (u32)img;
}

__global__ void k_reorder1(const u32* __restrict__ sidx, const float4* __restrict__ dbox,
                           const float* __restrict__ dgoff, const u32* __restrict__ dimg,
                           float4* clean, float4* off, u32* simg) {
  int r = blockIdx.x * blockDim.x + threadIdx.x;
  if (r >= NSEL) return;
  u32 s = sidx[r];
  float4 b = dbox[s];
  float g = dgoff[s];
  clean[r] = b;
  off[r] = make_float4(b.x + g, b.y + g, b.z + g, b.w + g);
  simg[r] = dimg[s];
}

// greedy NMS: single workgroup, boxes + flags in LDS (fits CDNA5's 320KB)
__global__ void __launch_bounds__(1024) k_nms(const float4* __restrict__ boxes,
                                              const u32* __restrict__ validInit,
                                              int n, float thr, u32* __restrict__ keep) {
  extern __shared__ char smem[];
  float4* bb = (float4*)smem;
  u32* kp = (u32*)(bb + n);
  for (int i = threadIdx.x; i < n; i += blockDim.x) {
    bb[i] = boxes[i];
    kp[i] = validInit ? validInit[i] : 1u;
  }
  __syncthreads();
  for (int i = 0; i < n - 1; ++i) {
    __syncthreads();
    if (!kp[i]) continue;
    float4 bi = bb[i];
    for (int j = i + 1 + threadIdx.x; j < n; j += blockDim.x) {
      if (!kp[j]) continue;
      if (iou_box(bi, bb[j]) > thr) kp[j] = 0u;
    }
  }
  __syncthreads();
  for (int i = threadIdx.x; i < n; i += blockDim.x) keep[i] = kp[i];
}

__global__ void k_props(const u32* keep, const u32* simg, const float4* clean,
                        float4* props, u32* valid) {
  if (threadIdx.x || blockIdx.x) return;
  int c0 = 0, c1 = 0;
  for (int r = 0; r < NSEL; ++r) {
    if (!keep[r]) continue;
    int im = (int)simg[r];
    int* c = im ? &c1 : &c0;
    if (*c >= 1000) continue;
    int slot = im * 1000 + *c;
    props[slot] = clean[r];
    valid[slot] = 1u;
    (*c)++;
  }
}

__global__ void k_roilvl(const float4* props, int* roilvl) {
  int i = blockIdx.x * blockDim.x + threadIdx.x;
  if (i >= NPROP) return;
  float4 b = props[i];
  float area = fmaxf((b.z - b.x) * (b.w - b.y), 1e-6f);
  float lf = floorf(4.f + log2f(sqrtf(area) / 224.f));
  lf = fminf(fmaxf(lf, 2.f), 5.f);
  roilvl[i] = (int)lf - 2;
}

// ---------------------------------------------------------------------------
// RoI-align (7x7 cells, 2x2 samples) -> bf16 features laid out as fc1's A
// ---------------------------------------------------------------------------
__global__ void __launch_bounds__(256) k_roialign(
    const float* f0, const float* f1, const float* f2, const float* f3,
    const float4* __restrict__ props, const int* __restrict__ roilvl,
    u16* __restrict__ feat) {
  int bx = blockIdx.x;
  int roi = bx / 49, cell = bx % 49;
  int py = cell / 7, px = cell % 7;
  int l = roilvl[roi];
  const float* fm = (l == 0) ? f0 : (l == 1) ? f1 : (l == 2) ? f2 : f3;
  int H = c_H[l], W = c_W[l];
  int img = roi / 1000;
  float sc = 1.f / (float)c_STRIDE[l];
  float4 rb = props[roi];
  float x1 = rb.x * sc - 0.5f, y1 = rb.y * sc - 0.5f;
  float bw = (rb.z * sc - 0.5f - x1) * (1.f / 7.f);
  float bh = (rb.w * sc - 0.5f - y1) * (1.f / 7.f);
  int sy0[4], sx0[4], sy1[4], sx1[4], oo[4];
  float wly[4], wlx[4];
#pragma unroll
  for (int t = 0; t < 4; ++t) {
    int sy = t >> 1, sx = t & 1;
    float gy = ((float)(2 * py + sy) + 0.5f) * 0.5f;
    float gx = ((float)(2 * px + sx) + 0.5f) * 0.5f;
    float ys = y1 + gy * bh, xs = x1 + gx * bw;
    oo[t] = (ys < -1.f) || (ys > (float)H) || (xs < -1.f) || (xs > (float)W);
    float y = fminf(fmaxf(ys, 0.f), (float)H - 1.f);
    float x = fminf(fmaxf(xs, 0.f), (float)W - 1.f);
    int iy0 = (int)floorf(y), ix0 = (int)floorf(x);
    sy0[t] = iy0; sx0[t] = ix0;
    sy1[t] = min(iy0 + 1, H - 1); sx1[t] = min(ix0 + 1, W - 1);
    wly[t] = y - (float)iy0; wlx[t] = x - (float)ix0;
  }
  size_t ib = (size_t)img * 256 * H * W;
  for (int c = threadIdx.x; c < 256; c += blockDim.x) {
    const float* f = fm + ib + (size_t)c * H * W;
    float acc = 0.f;
#pragma unroll
    for (int t = 0; t < 4; ++t) {
      if (oo[t]) continue;
      float ly = wly[t], lx = wlx[t];
      acc += f[sy0[t] * W + sx0[t]] * (1.f - ly) * (1.f - lx)
           + f[sy0[t] * W + sx1[t]] * (1.f - ly) * lx
           + f[sy1[t] * W + sx0[t]] * ly * (1.f - lx)
           + f[sy1[t] * W + sx1[t]] * ly * lx;
    }
    acc *= 0.25f;
    feat[(size_t)roi * FEATK + c * 49 + cell] = f2bf(acc);
  }
}

// ---------------------------------------------------------------------------
// Stage-2 head: softmax+mask, decode2, final outputs
// ---------------------------------------------------------------------------
__global__ void k_softmax_mask(const float* __restrict__ clsf, const u32* __restrict__ valid,
                               float* __restrict__ sm) {
  int i = blockIdx.x * blockDim.x + threadIdx.x;
  if (i >= NPROP) return;
  const float* lg = clsf + (size_t)i * 96;
  float mx = lg[0];
  for (int c = 1; c < 81; ++c) mx = fmaxf(mx, lg[c]);
  float sum = 0.f;
  for (int c = 0; c < 81; ++c) sum += expf(lg[c] - mx);
  float inv = 1.f / sum;
  u32 ok = valid[i];
  for (int c = 0; c < 80; ++c) {
    float s = expf(lg[c] - mx) * inv;
    sm[(size_t)i * 80 + c] = (s > 0.05f && ok) ? s : -1.f;
  }
}

__global__ void k_decode2(const float* __restrict__ top_s, const u32* __restrict__ cidx,
                          const float* __restrict__ breg, const float4* __restrict__ props,
                          const int* imsizes, float4* b2, float4* b2off,
                          u32* im2, u32* cls2, u32* valid2) {
  int i = blockIdx.x * blockDim.x + threadIdx.x;
  if (i >= 4096) return;
  float v = top_s[i];
  u32 ci = cidx[i];
  u32 rdx = ci / 80u, cl = ci % 80u;
  const float* rg = breg + (size_t)rdx * 320 + cl * 4;
  float4 p = props[rdx];
  float w = p.z - p.x, h = p.w - p.y;
  float cx = p.x + 0.5f * w + rg[0] * 0.1f * w;
  float cy = p.y + 0.5f * h + rg[1] * 0.1f * h;
  float ww = w * expf(fminf(rg[2] * 0.2f, BBOX_CLAMP));
  float hh = h * expf(fminf(rg[3] * 0.2f, BBOX_CLAMP));
  int im = (int)(rdx / 1000u);
  float Wc = (float)imsizes[im * 2 + 1], Hc = (float)imsizes[im * 2 + 0];
  float x1 = fminf(fmaxf(cx - 0.5f * ww, 0.f), Wc);
  float x2 = fminf(fmaxf(cx + 0.5f * ww, 0.f), Wc);
  float y1 = fminf(fmaxf(cy - 0.5f * hh, 0.f), Hc);
  float y2 = fminf(fmaxf(cy + 0.5f * hh, 0.f), Hc);
  float g = (float)(im * 80 + (int)cl) * 4096.f;
  b2[i] = make_float4(x1, y1, x2, y2);
  b2off[i] = make_float4(x1 + g, y1 + g, x2 + g, y2 + g);
  im2[i] = (u32)im;
  cls2[i] = cl;
  valid2[i] = (v > 0.f) ? 1u : 0u;
}

__global__ void k_final(const u32* keep2, const float4* b2, const float* top_s,
                        const u32* im2, const u32* cls2, float* out) {
  if (threadIdx.x || blockIdx.x) return;
  for (int i = 0; i < 1000; ++i) out[i] = 0.f;       // boxes + scores
  for (int i = 1000; i < 1200; ++i) out[i] = -1.f;   // classes
  int c0 = 0, c1 = 0;
  for (int r = 0; r < 4096; ++r) {
    if (!keep2[r]) continue;
    int im = (int)im2[r];
    int* c = im ? &c1 : &c0;
    if (*c >= 100) continue;
    int s = im * 100 + *c;
    float4 b = b2[r];
    out[s * 4 + 0] = b.x; out[s * 4 + 1] = b.y;
    out[s * 4 + 2] = b.z; out[s * 4 + 3] = b.w;
    out[800 + s] = top_s[r];
    out[1000 + s] = (float)cls2[r];
    (*c)++;
  }
}

// ---------------------------------------------------------------------------
// Host orchestration
// ---------------------------------------------------------------------------
struct SelScratch { u32* hist; u32* cnt; float* cv; u32* ci; float* sv; u32* si; };

static void run_select(const float* vals, int n, int k, u32* selIdx, float* selVal,
                       const SelScratch& S, hipStream_t st) {
  k_zero<<<4, 256, 0, st>>>(S.hist, 1024);
  k_zero<<<1, 32, 0, st>>>(S.cnt, 4);
  k_hist<<<(n + 255) / 256, 256, 0, st>>>(vals, n, S.hist);
  k_thresh<<<1, 1, 0, st>>>(S.hist, k, S.cnt);
  k_compact<<<(n + 255) / 256, 256, 0, st>>>(vals, n, S.cnt, S.cv, S.ci, 1);
  k_compact<<<(n + 255) / 256, 256, 0, st>>>(vals, n, S.cnt, S.cv, S.ci, 0);
  k_sort16k<<<1, 1024, SEL_CAP * 8, st>>>(S.cv, S.ci, S.cnt, 0, S.sv, S.si);
  k_take<<<(k + 255) / 256, 256, 0, st>>>(S.sv, S.si, k, n, selIdx, selVal);
}

extern "C" void kernel_launch(void* const* d_in, const int* in_sizes, int n_in,
                              void* d_out, int out_size, void* d_ws, size_t ws_size,
                              hipStream_t stream) {
  (void)in_sizes; (void)n_in; (void)out_size;
  static const int hHW[5]   = {26624, 6656, 1664, 416, 104};
  static const int hCum[6]  = {0, 26624, 33280, 34944, 35360, 35464};
  static const int hK[5]    = {1000, 1000, 1000, 1000, 312};
  static const int hCumK[6] = {0, 1000, 2000, 3000, 4000, 4312};

  const float* fmap[5]; for (int i = 0; i < 5; ++i) fmap[i] = (const float*)d_in[i];
  const float* pri[5];  for (int i = 0; i < 5; ++i) pri[i]  = (const float*)d_in[5 + i];
  const int* imsizes = (const int*)d_in[10];
  const float* rpn_conv_w = (const float*)d_in[11];
  const float* rpn_conv_b = (const float*)d_in[12];
  const float* rpn_log_w  = (const float*)d_in[13];
  const float* rpn_log_b  = (const float*)d_in[14];
  const float* rpn_reg_w  = (const float*)d_in[15];
  const float* rpn_reg_b  = (const float*)d_in[16];
  const float* fc1_w = (const float*)d_in[17];
  const float* fc1_b = (const float*)d_in[18];
  const float* fc2_w = (const float*)d_in[19];
  const float* fc2_b = (const float*)d_in[20];
  const float* cls_w = (const float*)d_in[21];
  const float* cls_b = (const float*)d_in[22];
  const float* breg_w = (const float*)d_in[23];
  const float* breg_b = (const float*)d_in[24];

  // workspace carve-up (256B aligned)
  char* base = (char*)d_ws;
  size_t off = 0;
  auto alloc = [&](size_t bytes) -> void* {
    size_t o = (off + 255) & ~(size_t)255;
    off = o + bytes;
    return (void*)(base + o);
  };
  u32*  fmapP  = (u32*)alloc(9078784ull * 4);
  u32*  convWp = (u32*)alloc(294912ull * 4);
  float* logits = (float*)alloc(2ull * LOGN * 4);
  float* regs   = (float*)alloc(2ull * LOGN * 16);
  u16* fc1wb  = (u16*)alloc(1024ull * FEATK * 2);
  u16* fc2wb  = (u16*)alloc(1024ull * 1024 * 2);
  u16* clswb  = (u16*)alloc(96ull * 1024 * 2);
  u16* bregwb = (u16*)alloc(320ull * 1024 * 2);
  u16* featb  = (u16*)alloc((size_t)NPROP * FEATK * 2);
  u16* h1b    = (u16*)alloc((size_t)NPROP * 1024 * 2);
  u16* h2b    = (u16*)alloc((size_t)NPROP * 1024 * 2);
  float* clsf  = (float*)alloc((size_t)NPROP * 96 * 4);
  float* bregf = (float*)alloc((size_t)NPROP * 320 * 4);
  float* smv   = (float*)alloc(160000ull * 4);
  u32*  sel    = (u32*)alloc(NSEL * 4);
  float4* dbox = (float4*)alloc(NSEL * 16);
  float* dscore = (float*)alloc(NSEL * 4);
  float* dgoff  = (float*)alloc(NSEL * 4);
  u32*  dimg   = (u32*)alloc(NSEL * 4);
  float4* sclean = (float4*)alloc(NSEL * 16);
  float4* soff   = (float4*)alloc(NSEL * 16);
  u32*  simg   = (u32*)alloc(NSEL * 4);
  u32*  keep1  = (u32*)alloc(NSEL * 4);
  float4* props = (float4*)alloc(NPROP * 16);
  u32*  rvalid = (u32*)alloc(NPROP * 4);
  int*  rlvl   = (int*)alloc(NPROP * 4);
  float* tops  = (float*)alloc(4096 * 4);
  u32*  cidx   = (u32*)alloc(4096 * 4);
  float4* b2   = (float4*)alloc(4096 * 16);
  float4* b2off = (float4*)alloc(4096 * 16);
  u32* im2 = (u32*)alloc(4096 * 4);
  u32* cls2 = (u32*)alloc(4096 * 4);
  u32* valid2 = (u32*)alloc(4096 * 4);
  u32* keep2 = (u32*)alloc(4096 * 4);
  SelScratch S;
  S.hist = (u32*)alloc(1024 * 4);
  S.cnt  = (u32*)alloc(8 * 4);
  S.cv   = (float*)alloc(SEL_CAP * 4);
  S.ci   = (u32*)alloc(SEL_CAP * 4);
  S.sv   = (float*)alloc(SEL_CAP * 4);
  S.si   = (u32*)alloc(SEL_CAP * 4);
  if (off > ws_size) return;  // workspace too small (needs ~135MB)

  // ---- pack fp32 -> bf16 ----
  k_pack_fmap<<<(9078784 + 255) / 256, 256, 0, stream>>>(
      fmap[0], fmap[1], fmap[2], fmap[3], fmap[4], fmapP, 9078784);
  k_pack_convw<<<(294912 + 255) / 256, 256, 0, stream>>>(rpn_conv_w, convWp);
  k_pack_mat<<<(1024 * FEATK + 255) / 256, 256, 0, stream>>>(fc1_w, fc1wb, 1024, FEATK, 1024);
  k_pack_mat<<<(1024 * 1024 + 255) / 256, 256, 0, stream>>>(fc2_w, fc2wb, 1024, 1024, 1024);
  k_pack_mat<<<(96 * 1024 + 255) / 256, 256, 0, stream>>>(cls_w, clswb, 81, 1024, 96);
  k_pack_mat<<<(320 * 1024 + 255) / 256, 256, 0, stream>>>(breg_w, bregwb, 320, 1024, 320);

  // ---- RPN fused conv + heads (WMMA, async-LDS double buffered) ----
  size_t convShm = (2 * 4096 + 2 * 1024) * 4 + (64 * 256 + 15 * 256 + 256) * 4;
  k_rpn_conv<<<1110, 256, convShm, stream>>>(fmapP, convWp, rpn_conv_b,
      rpn_log_w, rpn_log_b, rpn_reg_w, rpn_reg_b, logits, regs);

  // ---- per-(image,level) top-k on logits ----
  for (int img = 0; img < 2; ++img)
    for (int lvl = 0; lvl < 5; ++lvl)
      run_select(logits + (size_t)img * LOGN + 3 * hCum[lvl], 3 * hHW[lvl], hK[lvl],
                 sel + img * DTOT + hCumK[lvl], nullptr, S, stream);

  // ---- decode, global sort, NMS1, proposal compaction ----
  k_decode1<<<(NSEL + 255) / 256, 256, 0, stream>>>(logits, regs, sel,
      pri[0], pri[1], pri[2], pri[3], pri[4], imsizes, dbox, dscore, dgoff, dimg);
  k_sort16k<<<1, 1024, SEL_CAP * 8, stream>>>(dscore, nullptr, nullptr, NSEL, S.sv, S.si);
  k_reorder1<<<(NSEL + 255) / 256, 256, 0, stream>>>(S.si, dbox, dgoff, dimg,
                                                     sclean, soff, simg);
  k_nms<<<1, 1024, (size_t)NSEL * 20, stream>>>(soff, nullptr, NSEL, 0.7f, keep1);
  k_zero<<<(NPROP * 4 + 255) / 256, 256, 0, stream>>>((u32*)props, NPROP * 4);
  k_zero<<<(NPROP + 255) / 256, 256, 0, stream>>>(rvalid, NPROP);
  k_props<<<1, 1, 0, stream>>>(keep1, simg, sclean, props, rvalid);
  k_roilvl<<<(NPROP + 255) / 256, 256, 0, stream>>>(props, rlvl);

  // ---- RoI align -> bf16 features ----
  k_roialign<<<NPROP * 49, 256, 0, stream>>>(fmap[0], fmap[1], fmap[2], fmap[3],
                                             props, rlvl, featb);

  // ---- FC head GEMMs (WMMA bf16) ----
  {
    int Mt = NPROP / 16;
    int Ng, waves;
    Ng = 1024 / 64; waves = Mt * Ng;
    k_gemm<<<(waves + 3) / 4, 128, 0, stream>>>(featb, fc1wb, fc1_b, NPROP, FEATK,
                                                1024, 1024, 1, nullptr, h1b);
    k_gemm<<<(waves + 3) / 4, 128, 0, stream>>>(h1b, fc2wb, fc2_b, NPROP, 1024,
                                                1024, 1024, 1, nullptr, h2b);
    Ng = (96 + 63) / 64; waves = Mt * Ng;
    k_gemm<<<(waves + 3) / 4, 128, 0, stream>>>(h2b, clswb, cls_b, NPROP, 1024,
                                                96, 81, 0, clsf, nullptr);
    Ng = (320 + 63) / 64; waves = Mt * Ng;
    k_gemm<<<(waves + 3) / 4, 128, 0, stream>>>(h2b, bregwb, breg_b, NPROP, 1024,
                                                320, 320, 0, bregf, nullptr);
  }

  // ---- softmax+mask, top-4096, decode2, NMS2, final outputs ----
  k_softmax_mask<<<(NPROP + 255) / 256, 256, 0, stream>>>(clsf, rvalid, smv);
  run_select(smv, 160000, 4096, cidx, tops, S, stream);
  k_decode2<<<16, 256, 0, stream>>>(tops, cidx, bregf, props, imsizes,
                                    b2, b2off, im2, cls2, valid2);
  k_nms<<<1, 1024, 4096ull * 20, stream>>>(b2off, valid2, 4096, 0.5f, keep2);
  k_final<<<1, 1, 0, stream>>>(keep2, b2, tops, im2, cls2, (float*)d_out);
}
